// InterRowColSelfAttention_32658931319218
// MI455X (gfx1250) — compile-verified, matrix operand order checked by reading, not run on hardware
//
#include <hip/hip_runtime.h>
#include <hip/hip_bf16.h>
#include <math.h>

typedef __attribute__((ext_vector_type(16))) _Float16 v16h;
typedef __attribute__((ext_vector_type(8)))  _Float16 v8h;
typedef __attribute__((ext_vector_type(8)))  float    v8f;

// ---------------------------------------------------------------------------
// CDNA5 WMMA helpers (v_wmma_f32_16x16x32_f16, wave32)
// A (16x32 f16): lane L -> row M = L&15 ; element i -> K = i + 8*(L>>4) + (i>=8 ? 8 : 0)
// B (32x16 f16): lane L -> col N = L&15 ; element i -> K = same map
// C/D (16x16 f32): lane L -> col N = L&15 ; VGPR r -> row M = r + 8*(L>>4)
// kmap gives two contiguous 8-half runs per lane -> compiler emits ds_load_b128.
// ---------------------------------------------------------------------------
__device__ __forceinline__ int kmap(int i, int grp) {
  return i + 8 * grp + ((i >= 8) ? 8 : 0);
}

// Fragment gather from row-major [R][K] storage where R maps to the lane axis
// (used for A, and for B when the matrix is stored transposed as [N][K]).
__device__ __forceinline__ v16h frag_nk_f16(const _Float16* s, int r0, int k0, int ld) {
  const int lane = threadIdx.x & 31;
  const int row  = r0 + (lane & 15);
  const int grp  = lane >> 4;
  v16h f;
#pragma unroll
  for (int i = 0; i < 16; ++i) f[i] = s[row * ld + k0 + kmap(i, grp)];
  return f;
}

// Same gather from global f32 (weights), converting to f16.
__device__ __forceinline__ v16h frag_nk_f32g(const float* __restrict__ g, int r0, int k0, int ld) {
  const int lane = threadIdx.x & 31;
  const int row  = r0 + (lane & 15);
  const int grp  = lane >> 4;
  v16h f;
#pragma unroll
  for (int i = 0; i < 16; ++i) f[i] = (_Float16)g[row * ld + k0 + kmap(i, grp)];
  return f;
}

// True-B fragment from row-major [K][N] storage.
__device__ __forceinline__ v16h frag_kn_f16(const _Float16* s, int k0, int n0, int ld) {
  const int lane = threadIdx.x & 31;
  const int col  = n0 + (lane & 15);
  const int grp  = lane >> 4;
  v16h f;
#pragma unroll
  for (int i = 0; i < 16; ++i) f[i] = s[(k0 + kmap(i, grp)) * ld + col];
  return f;
}

__device__ __forceinline__ v8f wmma16(v16h a, v16h b, v8f c) {
  return __builtin_amdgcn_wmma_f32_16x16x32_f16(false, a, false, b, (short)0, c, false, false);
}

__device__ __forceinline__ void atomicMaxF(float* addr, float val) {
  unsigned int* ua = (unsigned int*)addr;
  unsigned int old = __float_as_uint(*addr);
  while (__uint_as_float(old) < val) {
    unsigned int assumed = old;
    old = atomicCAS(ua, assumed, __float_as_uint(val));
    if (old == assumed) break;
  }
}

// ---------------------------------------------------------------------------
// Kernel 0: init workspace accumulators (ws is poisoned by the harness)
// ---------------------------------------------------------------------------
__global__ void init_ws_kernel(float* __restrict__ gsum, float* __restrict__ gmax) {
  int i = blockIdx.x * blockDim.x + threadIdx.x;
  if (i < 1024) {
    gsum[i] = 0.0f;
    gmax[i] = -3.402823466e38f;
  }
}

// ---------------------------------------------------------------------------
// Kernel 1: per-(b,w) column attention, fused QKV + scores + softmax + AV +
//           gated residual + per-channel sum/max partial reduction.
// Grid: 1024 blocks (B*W), 256 threads (8 wave32).
// ---------------------------------------------------------------------------
__global__ __launch_bounds__(256) void fused_attn_kernel(
    const float* __restrict__ x, const float* __restrict__ proc,
    const float* __restrict__ Wq, const float* __restrict__ bq,
    const float* __restrict__ Wk, const float* __restrict__ bk,
    const float* __restrict__ Wv, const float* __restrict__ bv,
    const float* __restrict__ pos, const float* __restrict__ gamma,
    float* __restrict__ out, float* __restrict__ gsum, float* __restrict__ gmax)
{
  constexpr int C = 128, H = 128, W = 128, QK = 64;
  constexpr int SPS = 136;  // stride for P / scores [h][c] & [h][g] (pad vs 64 banks)
  constexpr int SQS = 72;   // stride for Q/K [h][q]
  constexpr int SVS = 136;  // stride for V [g][c]

  __shared__ _Float16 sPS[H * SPS];  // phase1: P[h][c] f16 ; phase2+: scores/attn[h][g]
  __shared__ _Float16 sQ[H * SQS];
  __shared__ _Float16 sK[H * SQS];
  __shared__ _Float16 sV[H * SVS];
  __shared__ float psum[C];
  __shared__ float pmax[C];

  const int b    = blockIdx.x >> 7;
  const int w    = blockIdx.x & 127;
  const int tid  = threadIdx.x;
  const int wave = tid >> 5;
  const int lane = tid & 31;
  const int lgrp = lane >> 4;   // 0 or 1: which half-wave
  const int l15  = lane & 15;

  // ---- per-wave GEMM-strip setup; cache weight B-fragments in registers
  //      BEFORE the staging barrier so the (L2-resident) weight gathers
  //      overlap the strided HBM staging of the processed column ----
  const int    qk_n0 = (wave & 3) * 16;
  const float* Wqk   = (wave < 4) ? Wq : Wk;
  const float* bqk   = (wave < 4) ? bq : bk;
  _Float16*    sQK   = (wave < 4) ? sQ : sK;
  const int    v_n0  = wave * 16;

  v16h bfQK[4], bfV[4];
#pragma unroll
  for (int kk = 0; kk < 4; ++kk) {
    bfQK[kk] = frag_nk_f32g(Wqk, qk_n0, kk * 32, C);   // [N][K] storage
    bfV[kk]  = frag_nk_f32g(Wv,  v_n0,  kk * 32, C);
  }

  // ---- stage processed[:, :, w] column -> LDS f16 [h][c] ----
  for (int idx = tid; idx < H * C; idx += 256) {
    int h = idx >> 7, c = idx & 127;
    sPS[h * SPS + c] = (_Float16)proc[(((size_t)b * C + c) * H + h) * W + w];
  }
  __syncthreads();

  // ============ fused Q/K/V projections, one sweep over M tiles ============
  // Waves 0-3 own Q n-strips 0-3; waves 4-7 own K n-strips 0-3; all waves own
  // one V n-strip.  Each A fragment feeds two WMMAs (no D->A/B hazard).
  {
    const int    qkcol  = qk_n0 + l15;
    const int    vcol   = v_n0 + l15;
    const float  qkbias = bqk[qkcol];
    const float  vbias  = bv[vcol];
    const float* posc   = pos + qkcol * H;   // pos_embed[0][q][h][0]

    for (int mt = 0; mt < 8; ++mt) {
      const int m0 = mt * 16;
      v8f accQK = {}, accV = {};
#pragma unroll
      for (int kk = 0; kk < 4; ++kk) {
        v16h a = frag_nk_f16(sPS, m0, kk * 32, SPS);
        accQK = wmma16(a, bfQK[kk], accQK);
        accV  = wmma16(a, bfV[kk],  accV);
      }
#pragma unroll
      for (int r = 0; r < 8; ++r) {
        const int row = m0 + r + 8 * lgrp;
        sQK[row * SQS + qkcol] = (_Float16)(accQK[r] + qkbias + posc[row]);
        sV[row * SVS + vcol]   = (_Float16)(accV[r] + vbias);
      }
    }
  }
  __syncthreads();  // P dead after this point; sPS becomes the scores buffer

  // ============ scores: S[h][g] = (1/8) * sum_q Q[h][q] * K[g][q] ============
  {
    const int n0 = wave * 16;
    v16h bf[2];
#pragma unroll
    for (int kk = 0; kk < 2; ++kk) bf[kk] = frag_nk_f16(sK, n0, kk * 32, SQS);
    const int col = n0 + l15;
    for (int mt = 0; mt < 8; ++mt) {
      const int m0 = mt * 16;
      v8f acc = {};
#pragma unroll
      for (int kk = 0; kk < 2; ++kk)
        acc = wmma16(frag_nk_f16(sQ, m0, kk * 32, SQS), bf[kk], acc);
#pragma unroll
      for (int r = 0; r < 8; ++r) {
        const int row = m0 + r + 8 * lgrp;
        sPS[row * SPS + col] = (_Float16)(acc[r] * 0.125f);
      }
    }
  }

  // Prefetch the x residual column for this wave's channel strip while the
  // scores settle (global_prefetch_b8; strided lines, partial coverage).
  {
    const int c = wave * 16 + l15;
    for (int j = 0; j < 16; ++j) {
      const int h = lgrp + 8 * j;
      __builtin_prefetch(&x[(((size_t)b * C + c) * H + h) * W + w], 0, 0);
    }
  }
  __syncthreads();

  // ============ softmax over g (one row per thread, threads 0..127) ============
  if (tid < H) {
    v8h* rowv = (v8h*)&sPS[tid * SPS];   // row base is 16B aligned (272B stride)
    float mx = -3.402823466e38f;
#pragma unroll 4
    for (int ch = 0; ch < 16; ++ch) {
      v8h v = rowv[ch];
#pragma unroll
      for (int j = 0; j < 8; ++j) mx = fmaxf(mx, (float)v[j]);
    }
    float sum = 0.0f;
#pragma unroll 4
    for (int ch = 0; ch < 16; ++ch) {
      v8h v = rowv[ch];
#pragma unroll
      for (int j = 0; j < 8; ++j) {
        float e = expf((float)v[j] - mx);
        sum += e;
        v[j] = (_Float16)e;
      }
      rowv[ch] = v;
    }
    const float inv = 1.0f / sum;
#pragma unroll 4
    for (int ch = 0; ch < 16; ++ch) {
      v8h v = rowv[ch];
#pragma unroll
      for (int j = 0; j < 8; ++j) v[j] = (_Float16)((float)v[j] * inv);
      rowv[ch] = v;
    }
  }
  __syncthreads();

  // ===== O[h][c] = sum_g attn[h][g] * V[g][c]; gated residual; channel stats =====
  {
    const float gsig = 1.0f / (1.0f + expf(-gamma[0]));
    const int   n0   = wave * 16;
    const int   c    = n0 + l15;          // fixed channel per lane for all M tiles
    v16h bf[4];
#pragma unroll
    for (int kk = 0; kk < 4; ++kk) bf[kk] = frag_kn_f16(sV, kk * 32, n0, SVS);

    float lsum = 0.0f, lmax = -3.402823466e38f;
    const size_t cbase = (((size_t)b * C + c) * H) * W + w;

    for (int mt = 0; mt < 8; ++mt) {
      const int m0 = mt * 16;
      v8f acc = {};
#pragma unroll
      for (int kk = 0; kk < 4; ++kk)
        acc = wmma16(frag_nk_f16(sPS, m0, kk * 32, SPS), bf[kk], acc);
#pragma unroll
      for (int r = 0; r < 8; ++r) {
        const int h = m0 + r + 8 * lgrp;
        const size_t gi = cbase + (size_t)h * W;
        const float val = gsig * acc[r] + (1.0f - gsig) * x[gi];
        out[gi] = val;
        lsum += val;
        lmax = fmaxf(lmax, val);
      }
    }
    // lanes L and L+16 share the same channel: combine across the half-waves
    lsum += __shfl_xor(lsum, 16);
    lmax = fmaxf(lmax, __shfl_xor(lmax, 16));
    if (lgrp == 0) { psum[c] = lsum; pmax[c] = lmax; }
  }
  __syncthreads();

  // ---- one global atomic per channel per workgroup ----
  if (tid < C) {
    atomicAdd(&gsum[b * C + tid], psum[tid]);
    atomicMaxF(&gmax[b * C + tid], pmax[tid]);
  }
}

// ---------------------------------------------------------------------------
// Kernel 2: channel-attention gate. fc2 is linear, so
//   fc2(relu(fc1*avg)) + fc2(relu(fc1*mx)) = fc2(relu(s_avg) + relu(s_mx)).
// Grid: B blocks of C threads.
// ---------------------------------------------------------------------------
__global__ void channel_attn_kernel(const float* __restrict__ fc1,
                                    const float* __restrict__ fc2,
                                    const float* __restrict__ gsum,
                                    const float* __restrict__ gmax,
                                    float* __restrict__ ca) {
  constexpr int C = 128, HID = 8;
  __shared__ float avg[C], mx[C], hid[HID];
  const int b = blockIdx.x;
  const int c = threadIdx.x;
  avg[c] = gsum[b * C + c] * (1.0f / 16384.0f);  // / (H*W)
  mx[c]  = gmax[b * C + c];
  __syncthreads();
  if (c < HID) {
    float s1 = 0.0f, s2 = 0.0f;
    for (int k = 0; k < C; ++k) {
      const float wv = fc1[c * C + k];
      s1 += wv * avg[k];
      s2 += wv * mx[k];
    }
    hid[c] = fmaxf(s1, 0.0f) + fmaxf(s2, 0.0f);
  }
  __syncthreads();
  float y = 0.0f;
#pragma unroll
  for (int o = 0; o < HID; ++o) y += fc2[c * HID + o] * hid[o];
  ca[b * C + c] = 1.0f / (1.0f + expf(-y));
}

// ---------------------------------------------------------------------------
// Kernel 3: out[b,c,h,w] *= ca[b,c]  (float4 vectorized, grid-stride)
// ---------------------------------------------------------------------------
__global__ void scale_kernel(float* __restrict__ out, const float* __restrict__ ca) {
  const long long total4 = (long long)8 * 128 * 128 * 128 / 4;
  for (long long i = (long long)blockIdx.x * blockDim.x + threadIdx.x; i < total4;
       i += (long long)gridDim.x * blockDim.x) {
    float4 v = ((float4*)out)[i];
    const float s = ca[(i * 4) >> 14];  // H*W = 16384 elems per (b,c)
    v.x *= s; v.y *= s; v.z *= s; v.w *= s;
    ((float4*)out)[i] = v;
  }
}

// ---------------------------------------------------------------------------
extern "C" void kernel_launch(void* const* d_in, const int* in_sizes, int n_in,
                              void* d_out, int out_size, void* d_ws, size_t ws_size,
                              hipStream_t stream) {
  const float* x     = (const float*)d_in[0];
  const float* proc  = (const float*)d_in[1];
  const float* Wq    = (const float*)d_in[2];
  const float* bq    = (const float*)d_in[3];
  const float* Wk    = (const float*)d_in[4];
  const float* bk    = (const float*)d_in[5];
  const float* Wv    = (const float*)d_in[6];
  const float* bv    = (const float*)d_in[7];
  const float* pos   = (const float*)d_in[8];
  const float* gamma = (const float*)d_in[9];
  const float* fc1   = (const float*)d_in[10];
  const float* fc2   = (const float*)d_in[11];
  float* out = (float*)d_out;

  float* ws   = (float*)d_ws;
  float* gsum = ws;          // 1024 floats (B*C)
  float* gmax = ws + 1024;   // 1024 floats
  float* ca   = ws + 2048;   // 1024 floats

  init_ws_kernel<<<4, 256, 0, stream>>>(gsum, gmax);
  fused_attn_kernel<<<1024, 256, 0, stream>>>(x, proc, Wq, bq, Wk, bk, Wv, bv,
                                              pos, gamma, out, gsum, gmax);
  channel_attn_kernel<<<8, 128, 0, stream>>>(fc1, fc2, gsum, gmax, ca);
  scale_kernel<<<2048, 256, 0, stream>>>(out, ca);
}
